// MixtureOfExperts_55645596287145
// MI455X (gfx1250) — matrix operand order repaired and checked by reading, not compiled, not run on hardware
//
#include <hip/hip_runtime.h>
#include <hip/hip_bf16.h>

// ---------------------------------------------------------------------------
// MoE (shared SwiGLU + 8 routed experts, top-2) for MI455X / gfx1250.
// bf16 WMMA GEMMs, double-buffered async global->LDS panel staging (ASYNCcnt),
// 2x2 register blocking per wave.
// ---------------------------------------------------------------------------

typedef __bf16 bf16;
typedef __attribute__((ext_vector_type(16))) __bf16 v16bf;
typedef __attribute__((ext_vector_type(8)))  __bf16 v8bf;
typedef __attribute__((ext_vector_type(8)))  float  v8f;

#define D_MODEL 1024
#define D_FF    4096
#define N_TOK   4096     // B*T = 2*2048
#define N_EXP   8
// 8192 routed rows + up to 8*63 pad + 4096 shared rows, rounded to 64:
#define ROWS_CAP 12800
#define ROW_TILES (ROWS_CAP / 64)   // 200

// ctrl block ints: [0..7]=counts [8..16]=start (start[8]=shared base) [17..24]=cursor
#define C_COUNT  0
#define C_START  8
#define C_CURSOR 17

// ---------------------------------------------------------------------------
__global__ void k_zero_f32(float* __restrict__ p, int n) {
    int i = blockIdx.x * 256 + threadIdx.x;
    if (i < n) p[i] = 0.0f;
}

__global__ void k_zero_ctrl(int* __restrict__ ctrl) {
    if (threadIdx.x < 32) ctrl[threadIdx.x] = 0;
}

// ---------------------------------------------------------------------------
// Gate: logits = x @ gate_w, top-2 + softmax; also converts x -> bf16.
__global__ void k_gate(const float* __restrict__ x,
                       const float* __restrict__ gw,
                       bf16* __restrict__ xh,
                       int* __restrict__ gateSel,
                       float* __restrict__ gateW,
                       int* __restrict__ ctrl) {
    const int t   = blockIdx.x;
    const int tid = threadIdx.x;
    __shared__ float sp[128 * N_EXP];
    __shared__ float lg[N_EXP];

    const float* xr = x + (size_t)t * D_MODEL;
    float p[N_EXP];
#pragma unroll
    for (int e = 0; e < N_EXP; ++e) p[e] = 0.0f;
#pragma unroll
    for (int j = 0; j < 8; ++j) {
        int c   = tid * 8 + j;
        float v = xr[c];
        xh[(size_t)t * D_MODEL + c] = (bf16)v;
#pragma unroll
        for (int e = 0; e < N_EXP; ++e) p[e] += v * gw[c * N_EXP + e];
    }
#pragma unroll
    for (int e = 0; e < N_EXP; ++e) sp[tid * N_EXP + e] = p[e];
    __syncthreads();
    if (tid < N_EXP) {
        float s = 0.0f;
        for (int i = 0; i < 128; ++i) s += sp[i * N_EXP + tid];
        lg[tid] = s;
    }
    __syncthreads();
    if (tid == 0) {
        float best = -3.0e38f, second = -3.0e38f;
        int b0 = 0, b1 = 0;
#pragma unroll
        for (int e = 0; e < N_EXP; ++e) {
            float v = lg[e];
            if (v > best)        { second = best; b1 = b0; best = v; b0 = e; }
            else if (v > second) { second = v; b1 = e; }
        }
        float p0 = 1.0f / (1.0f + __expf(second - best));
        float p1 = 1.0f - p0;
        gateSel[2 * t]     = b0;  gateW[2 * t]     = p0;
        gateSel[2 * t + 1] = b1;  gateW[2 * t + 1] = p1;
        atomicAdd(&ctrl[C_COUNT + b0], 1);
        atomicAdd(&ctrl[C_COUNT + b1], 1);
    }
}

// ---------------------------------------------------------------------------
__global__ void k_scan(int* __restrict__ ctrl) {
    int cum = 0;
    for (int e = 0; e < N_EXP; ++e) {
        ctrl[C_START + e] = cum;
        cum += ((ctrl[C_COUNT + e] + 63) / 64) * 64;
        ctrl[C_CURSOR + e] = 0;
    }
    ctrl[C_START + N_EXP] = cum;
}

__global__ void k_fill(const int* __restrict__ ctrl,
                       int* __restrict__ tokenOf,
                       float* __restrict__ weightOf,
                       int* __restrict__ expertOf) {
    int r = blockIdx.x * 256 + threadIdx.x;
    if (r >= ROWS_CAP) return;
    int sharedStart = ctrl[C_START + N_EXP];
    int e = N_EXP, tok = 0;
    float w = 0.0f;
    if (r < sharedStart) {
        for (int i = 0; i < N_EXP; ++i)
            if (r >= ctrl[C_START + i] && r < ctrl[C_START + i + 1]) e = i;
    } else {
        int i = r - sharedStart;
        if (i < N_TOK) { tok = i; w = 1.0f; }
    }
    expertOf[r] = e;
    tokenOf[r]  = tok;
    weightOf[r] = w;
}

__global__ void k_scatter(int* __restrict__ ctrl,
                          const int* __restrict__ gateSel,
                          const float* __restrict__ gateW,
                          int* __restrict__ tokenOf,
                          float* __restrict__ weightOf) {
    int t = blockIdx.x * 256 + threadIdx.x;
    if (t >= N_TOK) return;
#pragma unroll
    for (int s = 0; s < 2; ++s) {
        int e    = gateSel[2 * t + s];
        int pos  = atomicAdd(&ctrl[C_CURSOR + e], 1);
        int idx  = ctrl[C_START + e] + pos;
        tokenOf[idx]  = t;
        weightOf[idx] = gateW[2 * t + s];
    }
}

// ---------------------------------------------------------------------------
// Transpose + convert f32 [R,C] -> bf16 [C,R]; blockIdx.z selects matrix.
__global__ void k_tcvt(const float* __restrict__ in, bf16* __restrict__ out,
                       int R, int C) {
    __shared__ float tile[32][33];
    size_t off = (size_t)blockIdx.z * R * C;
    int c0 = blockIdx.x * 32, r0 = blockIdx.y * 32;
    int tx = threadIdx.x, ty = threadIdx.y;  // (32, 8)
#pragma unroll
    for (int k = 0; k < 4; ++k)
        tile[ty + 8 * k][tx] = in[off + (size_t)(r0 + ty + 8 * k) * C + c0 + tx];
    __syncthreads();
#pragma unroll
    for (int k = 0; k < 4; ++k)
        out[off + (size_t)(c0 + ty + 8 * k) * R + r0 + tx] =
            (bf16)tile[tx][ty + 8 * k];
}

// ---------------------------------------------------------------------------
// CDNA5 async global -> LDS copy, 16 bytes per lane (ASYNCcnt tracked).
__device__ __forceinline__ void async_copy_b128(unsigned ldsOff,
                                                unsigned long long gaddr) {
    asm volatile("global_load_async_to_lds_b128 %0, %1, off"
                 :: "v"(ldsOff), "v"(gaddr) : "memory");
}
template <int N>
__device__ __forceinline__ void async_wait() {
    asm volatile("s_wait_asynccnt %0" :: "i"(N) : "memory");
}

// ---------------------------------------------------------------------------
// WMMA fragment loaders (contiguous k order, same for A and B, wave32), reading LDS panels.
__device__ __forceinline__ v16bf frag_a(const bf16* row, int h) {
    // lane(m=l&15,h=l>>4): elems 0..15 = K[16h..+16) contiguous
    return *(const v16bf*)(row + 16 * h);
}
__device__ __forceinline__ v16bf frag_b(const bf16* row, int h) {
    // lane(n=l&15,h=l>>4): elems 0..15 = K[16h..+16) contiguous
    return *(const v16bf*)(row + 16 * h);
}
__device__ __forceinline__ v8f wmma_bf16(v16bf a, v16bf b, v8f c) {
    return __builtin_amdgcn_wmma_f32_16x16x32_bf16(false, a, false, b,
                                                   (short)0, c, false, false);
}

// ---------------------------------------------------------------------------
// FFN1: act[r, f] = silu(x[t]@W1)[f] * (x[t]@W2)[f], gathered rows.
// Block 256 thr = 8 waves (wm 0..1, wn 0..3); block tile 64 rows x 128 f-cols;
// wave tile 32x32 per matrix (2x2 WMMA). K panel = 32 elems, DOUBLE-buffered
// async staging. Dynamic LDS: 2 x { A 4KB | B1 8KB | B2 8KB } = 40KB.
__global__ void k_ffn1(const bf16* __restrict__ xh,
                       const bf16* __restrict__ sw1t,
                       const bf16* __restrict__ sw2t,
                       const bf16* __restrict__ ew1t,
                       const bf16* __restrict__ ew2t,
                       const int* __restrict__ tokenOf,
                       const int* __restrict__ expertOf,
                       bf16* __restrict__ act) {
    extern __shared__ char smem_raw[];

    const int rB = blockIdx.x * 64;
    const int fB = blockIdx.y * 128;
    const int e  = expertOf[rB];
    const bf16* w1 = (e < N_EXP) ? ew1t + (size_t)e * D_FF * D_MODEL : sw1t;
    const bf16* w2 = (e < N_EXP) ? ew2t + (size_t)e * D_FF * D_MODEL : sw2t;

    const int t    = threadIdx.x;
    const int lane = t & 31;
    const int wid  = t >> 5;
    const int wm   = wid >> 2;   // 0..1
    const int wn   = wid & 3;    // 0..3
    const int m    = lane & 15;
    const int h    = lane >> 4;

    // async sources: panel = 32 K-elems = 64B/row -> 4 chunks/row, thread t
    // handles A chunk t (row t>>2) and B chunks t, t+256 (rows t>>2, t>>2+64).
    const int aRow = t >> 2;
    const int sub  = (t & 3) * 16;
    unsigned long long gA = (unsigned long long)(uintptr_t)xh +
        (unsigned long long)tokenOf[rB + aRow] * (D_MODEL * 2) + sub;
    unsigned long long gB1a = (unsigned long long)(uintptr_t)w1 +
        (unsigned long long)(fB + aRow) * (D_MODEL * 2) + sub;
    unsigned long long gB1b = gB1a + 64ull * (D_MODEL * 2);
    unsigned long long gB2a = (unsigned long long)(uintptr_t)w2 +
        (unsigned long long)(fB + aRow) * (D_MODEL * 2) + sub;
    unsigned long long gB2b = gB2a + 64ull * (D_MODEL * 2);

    auto stage = [&](unsigned bufBase, int kb) {
        async_copy_b128(bufBase + t * 16,                 gA   + kb);
        async_copy_b128(bufBase + 4096 + t * 16,          gB1a + kb);
        async_copy_b128(bufBase + 4096 + (t + 256) * 16,  gB1b + kb);
        async_copy_b128(bufBase + 12288 + t * 16,         gB2a + kb);
        async_copy_b128(bufBase + 12288 + (t + 256) * 16, gB2b + kb);
    };

    v8f acc1[2][2] = {}, acc2[2][2] = {};
    const int NP = D_MODEL / 32;   // 32 panels, 64 bytes of K each

    stage(0, 0);
    for (int kp = 0; kp < NP; ++kp) {
        if (kp + 1 < NP) {
            stage(((kp + 1) & 1) * 20480u, (kp + 1) * 64);
            async_wait<5>();          // panel kp landed (in-order completion)
        } else {
            async_wait<0>();
        }
        __syncthreads();

        const bf16* sA  = (const bf16*)(smem_raw + (kp & 1) * 20480);
        const bf16* sB1 = sA + 2048;   // +4096 B
        const bf16* sB2 = sA + 6144;   // +12288 B
        v16bf a0  = frag_a(sA + (wm * 32 + m) * 32, h);
        v16bf a1  = frag_a(sA + (wm * 32 + 16 + m) * 32, h);
        v16bf b10 = frag_b(sB1 + (wn * 32 + m) * 32, h);
        v16bf b11 = frag_b(sB1 + (wn * 32 + 16 + m) * 32, h);
        v16bf b20 = frag_b(sB2 + (wn * 32 + m) * 32, h);
        v16bf b21 = frag_b(sB2 + (wn * 32 + 16 + m) * 32, h);
        acc1[0][0] = wmma_bf16(a0, b10, acc1[0][0]);
        acc1[0][1] = wmma_bf16(a0, b11, acc1[0][1]);
        acc1[1][0] = wmma_bf16(a1, b10, acc1[1][0]);
        acc1[1][1] = wmma_bf16(a1, b11, acc1[1][1]);
        acc2[0][0] = wmma_bf16(a0, b20, acc2[0][0]);
        acc2[0][1] = wmma_bf16(a0, b21, acc2[0][1]);
        acc2[1][0] = wmma_bf16(a1, b20, acc2[1][0]);
        acc2[1][1] = wmma_bf16(a1, b21, acc2[1][1]);
        __syncthreads();
    }

    // --- epilogue: silu(h1)*h2 -> LDS -> coalesced bf16 stores ---------------
    bf16* sOut = (bf16*)smem_raw;  // 64 x 128 bf16 = 16KB (reuse buffers)
#pragma unroll
    for (int mi = 0; mi < 2; ++mi)
#pragma unroll
        for (int ni = 0; ni < 2; ++ni)
#pragma unroll
            for (int i = 0; i < 8; ++i) {
                float g  = acc1[mi][ni][i];
                float u  = acc2[mi][ni][i];
                float sv = (g / (1.0f + __expf(-g))) * u;
                int row = wm * 32 + mi * 16 + 8 * h + i;
                int col = wn * 32 + ni * 16 + m;
                sOut[row * 128 + col] = (bf16)sv;
            }
    __syncthreads();
    {
        int row = t >> 2;             // 4 threads per row
        int cb  = (t & 3) * 32;       // 32 elems per thread
        bf16* dst = act + (size_t)(rB + row) * D_FF + fB + cb;
        const bf16* src = sOut + row * 128 + cb;
#pragma unroll
        for (int j = 0; j < 4; ++j)
            *(v8bf*)(dst + j * 8) = *(const v8bf*)(src + j * 8);
    }
}

// ---------------------------------------------------------------------------
// FFN2: out[t] += w * (act[r] @ W3). Block tile 64 rows x 128 out-cols,
// wave tile 32x32 (2x2 WMMA). K = 4096 in 32-elem DOUBLE-buffered panels.
// Dynamic LDS: 2 x { A 4KB | B 8KB } = 24KB.
__global__ void k_ffn2(const bf16* __restrict__ act,
                       const bf16* __restrict__ sw3t,
                       const bf16* __restrict__ ew3t,
                       const int* __restrict__ tokenOf,
                       const float* __restrict__ weightOf,
                       const int* __restrict__ expertOf,
                       float* __restrict__ out) {
    extern __shared__ char smem_raw[];

    const int rB = blockIdx.x * 64;
    const int cB = blockIdx.y * 128;
    const int e  = expertOf[rB];
    const bf16* w3 = (e < N_EXP) ? ew3t + (size_t)e * D_MODEL * D_FF : sw3t;

    const int t    = threadIdx.x;
    const int lane = t & 31;
    const int wid  = t >> 5;
    const int wm   = wid >> 2;   // 0..1
    const int wn   = wid & 3;    // 0..3
    const int m    = lane & 15;
    const int h    = lane >> 4;

    const int aRow = t >> 2;
    const int sub  = (t & 3) * 16;
    unsigned long long gA = (unsigned long long)(uintptr_t)act +
        (unsigned long long)(rB + aRow) * (D_FF * 2) + sub;
    unsigned long long gBa = (unsigned long long)(uintptr_t)w3 +
        (unsigned long long)(cB + aRow) * (D_FF * 2) + sub;
    unsigned long long gBb = gBa + 64ull * (D_FF * 2);

    auto stage = [&](unsigned bufBase, int kb) {
        async_copy_b128(bufBase + t * 16,                gA  + kb);
        async_copy_b128(bufBase + 4096 + t * 16,         gBa + kb);
        async_copy_b128(bufBase + 4096 + (t + 256) * 16, gBb + kb);
    };

    v8f acc[2][2] = {};
    const int NP = D_FF / 32;   // 128 panels, 64 bytes of K each

    stage(0, 0);
    for (int kp = 0; kp < NP; ++kp) {
        if (kp + 1 < NP) {
            stage(((kp + 1) & 1) * 12288u, (kp + 1) * 64);
            async_wait<3>();
        } else {
            async_wait<0>();
        }
        __syncthreads();

        const bf16* sA = (const bf16*)(smem_raw + (kp & 1) * 12288);
        const bf16* sB = sA + 2048;    // +4096 B
        v16bf a0 = frag_a(sA + (wm * 32 + m) * 32, h);
        v16bf a1 = frag_a(sA + (wm * 32 + 16 + m) * 32, h);
        v16bf b0 = frag_b(sB + (wn * 32 + m) * 32, h);
        v16bf b1 = frag_b(sB + (wn * 32 + 16 + m) * 32, h);
        acc[0][0] = wmma_bf16(a0, b0, acc[0][0]);
        acc[0][1] = wmma_bf16(a0, b1, acc[0][1]);
        acc[1][0] = wmma_bf16(a1, b0, acc[1][0]);
        acc[1][1] = wmma_bf16(a1, b1, acc[1][1]);
        __syncthreads();
    }

#pragma unroll
    for (int mi = 0; mi < 2; ++mi)
#pragma unroll
        for (int i = 0; i < 8; ++i) {
            int row = rB + wm * 32 + mi * 16 + 8 * h + i;
            int tok = tokenOf[row];
            float w = weightOf[row];
#pragma unroll
            for (int ni = 0; ni < 2; ++ni) {
                int col = cB + wn * 32 + ni * 16 + m;
                atomicAdd(&out[(size_t)tok * D_MODEL + col],
                          w * acc[mi][ni][i]);
            }
        }
}

// ---------------------------------------------------------------------------
extern "C" void kernel_launch(void* const* d_in, const int* in_sizes, int n_in,
                              void* d_out, int out_size, void* d_ws, size_t ws_size,
                              hipStream_t stream) {
    (void)in_sizes; (void)n_in; (void)ws_size;
    const float* x   = (const float*)d_in[0];
    const float* sw1 = (const float*)d_in[1];
    const float* sw2 = (const float*)d_in[2];
    const float* sw3 = (const float*)d_in[3];
    const float* ew1 = (const float*)d_in[4];
    const float* ew2 = (const float*)d_in[5];
    const float* ew3 = (const float*)d_in[6];
    const float* gw  = (const float*)d_in[7];
    float* out = (float*)d_out;

    char* w = (char*)d_ws;
    auto carve = [&](size_t bytes) -> void* {
        void* p = (void*)w;
        w += (bytes + 511) & ~(size_t)511;
        return p;
    };
    bf16*  xh       = (bf16*) carve((size_t)N_TOK * D_MODEL * 2);
    int*   ctrl     = (int*)  carve(512);
    int*   gateSel  = (int*)  carve((size_t)N_TOK * 2 * 4);
    float* gateW    = (float*)carve((size_t)N_TOK * 2 * 4);
    int*   tokenOf  = (int*)  carve((size_t)ROWS_CAP * 4);
    float* weightOf = (float*)carve((size_t)ROWS_CAP * 4);
    int*   expertOf = (int*)  carve((size_t)ROWS_CAP * 4);
    bf16*  sw1t = (bf16*)carve((size_t)D_FF * D_MODEL * 2);
    bf16*  sw2t = (bf16*)carve((size_t)D_FF * D_MODEL * 2);
    bf16*  sw3t = (bf16*)carve((size_t)D_MODEL * D_FF * 2);
    bf16*  ew1t = (bf16*)carve((size_t)N_EXP * D_FF * D_MODEL * 2);
    bf16*  ew2t = (bf16*)carve((size_t)N_EXP * D_FF * D_MODEL * 2);
    bf16*  ew3t = (bf16*)carve((size_t)N_EXP * D_MODEL * D_FF * 2);
    bf16*  act  = (bf16*)carve((size_t)ROWS_CAP * D_FF * 2);

    k_zero_f32<<<(out_size + 255) / 256, 256, 0, stream>>>(out, out_size);
    k_zero_ctrl<<<1, 64, 0, stream>>>(ctrl);

    k_gate<<<N_TOK, 128, 0, stream>>>(x, gw, xh, gateSel, gateW, ctrl);

    k_scan<<<1, 1, 0, stream>>>(ctrl);
    k_fill<<<(ROWS_CAP + 255) / 256, 256, 0, stream>>>(ctrl, tokenOf, weightOf,
                                                       expertOf);
    k_scatter<<<(N_TOK + 255) / 256, 256, 0, stream>>>(ctrl, gateSel, gateW,
                                                       tokenOf, weightOf);

    dim3 tb(32, 8);
    k_tcvt<<<dim3(D_FF / 32, D_MODEL / 32, 1),     tb, 0, stream>>>(sw1, sw1t, D_MODEL, D_FF);
    k_tcvt<<<dim3(D_FF / 32, D_MODEL / 32, 1),     tb, 0, stream>>>(sw2, sw2t, D_MODEL, D_FF);
    k_tcvt<<<dim3(D_MODEL / 32, D_FF / 32, 1),     tb, 0, stream>>>(sw3, sw3t, D_FF, D_MODEL);
    k_tcvt<<<dim3(D_FF / 32, D_MODEL / 32, N_EXP), tb, 0, stream>>>(ew1, ew1t, D_MODEL, D_FF);
    k_tcvt<<<dim3(D_FF / 32, D_MODEL / 32, N_EXP), tb, 0, stream>>>(ew2, ew2t, D_MODEL, D_FF);
    k_tcvt<<<dim3(D_MODEL / 32, D_FF / 32, N_EXP), tb, 0, stream>>>(ew3, ew3t, D_FF, D_MODEL);

    // FFN1: block tile 64 x 128, LDS 40KB dynamic (2 x 20KB buffers)
    k_ffn1<<<dim3(ROW_TILES, D_FF / 128), 256, 40960, stream>>>(
        xh, sw1t, sw2t, ew1t, ew2t, tokenOf, expertOf, act);

    // FFN2: block tile 64 x 128, LDS 24KB dynamic (2 x 12KB buffers)
    k_ffn2<<<dim3(ROW_TILES, D_MODEL / 128), 256, 24576, stream>>>(
        act, sw3t, ew3t, tokenOf, weightOf, expertOf, out);
}
